// AttnBlock_78280073937092
// MI455X (gfx1250) — matrix operand order repaired and checked by reading, not compiled, not run on hardware
//
#include <hip/hip_runtime.h>

typedef __attribute__((ext_vector_type(16))) __bf16 v16bf;
typedef __attribute__((ext_vector_type(8)))  float  v8f;

#define TM 128
#define TN 128
#define TK 32
#define LDSS 40   // LDS row stride in bf16 elems (80B rows: 16B-aligned, bank-skewed)

// ---------------------------------------------------------------------------
// GroupNorm stats: one block per (b,g); group = 16 ch * 1024 px contiguous.
// ---------------------------------------------------------------------------
__global__ __launch_bounds__(256) void gn_stats_kernel(
    const float* __restrict__ x, float* __restrict__ stats) {
  const int grp = blockIdx.x;                 // b*32 + g, 512 total
  const float4* p4 = (const float4*)(x + (size_t)grp * 16384);
  float s = 0.f, ss = 0.f;
  for (int i = threadIdx.x; i < 4096; i += 256) {
    float4 v = p4[i];
    s  += v.x + v.y + v.z + v.w;
    ss += v.x*v.x + v.y*v.y + v.z*v.z + v.w*v.w;
  }
  __shared__ float rs[256], rq[256];
  const int tid = threadIdx.x;
  rs[tid] = s; rq[tid] = ss;
  __syncthreads();
  for (int off = 128; off > 0; off >>= 1) {
    if (tid < off) { rs[tid] += rs[tid + off]; rq[tid] += rq[tid + off]; }
    __syncthreads();
  }
  if (tid == 0) {
    float mean = rs[0] * (1.f / 16384.f);
    float var  = rq[0] * (1.f / 16384.f) - mean * mean;
    stats[2 * grp]     = mean;
    stats[2 * grp + 1] = rsqrtf(var + 1e-5f);
  }
}

// ---------------------------------------------------------------------------
// GN affine apply + f32->bf16, hn[B,C,T]
// ---------------------------------------------------------------------------
__global__ __launch_bounds__(256) void gn_apply_kernel(
    const float* __restrict__ x, const float* __restrict__ stats,
    const float* __restrict__ gamma, const float* __restrict__ beta,
    __bf16* __restrict__ hn) {
  size_t i4   = (size_t)blockIdx.x * 256 + threadIdx.x;
  size_t base = i4 * 4;                       // 4 consecutive px, same channel
  int c   = (int)((base >> 10) & 511);        // T = 1024
  int b   = (int)(base >> 19);                // C*T = 2^19
  int grp = b * 32 + (c >> 4);
  float mean = stats[2 * grp], rstd = stats[2 * grp + 1];
  float ga = gamma[c] * rstd;
  float be = beta[c] - mean * ga;
  float4 v = *(const float4*)(x + base);
  union { __bf16 h[4]; uint2 u; } o;
  o.h[0] = (__bf16)(v.x * ga + be);
  o.h[1] = (__bf16)(v.y * ga + be);
  o.h[2] = (__bf16)(v.z * ga + be);
  o.h[3] = (__bf16)(v.w * ga + be);
  *(uint2*)(hn + base) = o.u;
}

// ---------------------------------------------------------------------------
// Row softmax over S=1024; writes bf16 probs in place (front half of each f32
// row; bf16 row stride = 2*S elems so rows never overlap each other).
// ---------------------------------------------------------------------------
__global__ __launch_bounds__(256) void softmax_kernel(float* __restrict__ scores) {
  float* rp = scores + (size_t)blockIdx.x * 1024;
  const int tid = threadIdx.x;
  float4 v = ((const float4*)rp)[tid];        // all global reads happen here
  __shared__ float red[256];
  float m = fmaxf(fmaxf(v.x, v.y), fmaxf(v.z, v.w));
  red[tid] = m;
  __syncthreads();
  for (int off = 128; off > 0; off >>= 1) {
    if (tid < off) red[tid] = fmaxf(red[tid], red[tid + off]);
    __syncthreads();
  }
  float rowmax = red[0];
  __syncthreads();
  v.x = __expf(v.x - rowmax); v.y = __expf(v.y - rowmax);
  v.z = __expf(v.z - rowmax); v.w = __expf(v.w - rowmax);
  red[tid] = v.x + v.y + v.z + v.w;
  __syncthreads();
  for (int off = 128; off > 0; off >>= 1) {
    if (tid < off) red[tid] += red[tid + off];
    __syncthreads();
  }
  float inv = 1.f / red[0];
  union { __bf16 h[4]; uint2 u; } o;
  o.h[0] = (__bf16)(v.x * inv); o.h[1] = (__bf16)(v.y * inv);
  o.h[2] = (__bf16)(v.z * inv); o.h[3] = (__bf16)(v.w * inv);
  ((uint2*)rp)[tid] = o.u;                    // bytes [tid*8, tid*8+8): front 2KB
}

// ---------------------------------------------------------------------------
// Generic batched WMMA GEMM: D = scale*A@B (+bias[m]) (+res).  Double-buffered
// LDS (one barrier per K-step), block tile 128x128, 8 waves each owning 32x64
// (2x4 WMMA accumulators -> 8 v_wmma per K-step).  A: f32 (converted on load)
// or bf16.  B: bf16, optionally transposed source.  D: f32 or bf16, optional
// transposed store.  All dims multiples of tile sizes (true here).
// ---------------------------------------------------------------------------
__global__ __launch_bounds__(256) void gemm_wmma_kernel(
    const void* __restrict__ Ap, int a_f32, long sAb, int lda,
    const __bf16* __restrict__ Bp, int b_trans, long sBb, int ldb,
    void* __restrict__ Dp, int d_bf16, int d_trans, long sDb, int ldo,
    const float* __restrict__ bias,
    const float* __restrict__ res, long sRb, int ldr,
    float scale, int K) {
  __shared__ __bf16 As[2][TM * LDSS];
  __shared__ __bf16 Bt[2][TN * LDSS];   // stored transposed: Bt[n][k]

  const int tid  = threadIdx.x;
  const int b    = blockIdx.z;
  const int m0   = blockIdx.y * TM;
  const int n0   = blockIdx.x * TN;
  const int wid  = tid >> 5;
  const int lane = tid & 31;
  const int lhalf = lane >> 4;
  const int l16   = lane & 15;
  const int miw = (wid & 3) * 32;    // wave M offset within tile
  const int niw = (wid >> 2) * 64;   // wave N offset within tile

  union F  { unsigned u[8]; v16bf v; };
  union A8 { float f[8];    v8f   v; };
  union P8 { uint4 q; __bf16 h[8]; };
  A8 acc[2][4];
#pragma unroll
  for (int i = 0; i < 2; ++i)
#pragma unroll
    for (int j = 0; j < 4; ++j)
#pragma unroll
      for (int r = 0; r < 8; ++r) acc[i][j].f[r] = 0.f;

  const float*  Af = (const float*)Ap;
  const __bf16* Ab = (const __bf16*)Ap;
  const size_t aBase = (size_t)b * sAb;
  const size_t bBase = (size_t)b * sBb;

  // per-thread staging coordinates
  const int arow = tid >> 1, akb = (tid & 1) * 16;       // A: 128 rows x 32 k
  const int brow = tid >> 1, bkb = (tid & 1) * 16;       // B trans: 128 n x 32 k
  const int krow = tid >> 3, nb  = (tid & 7) * 16;       // B normal: 32 k x 128 n

  // ---- global -> regs for one K-slab ----
  auto load_regs = [&](int kk, P8& alo, P8& ahi, P8& blo, P8& bhi) {
    if (a_f32) {
      const float* s = Af + aBase + (size_t)(m0 + arow) * lda + kk + akb;
#pragma unroll
      for (int i = 0; i < 2; ++i) {
        float4 f0 = *(const float4*)(s + 8 * i);
        float4 f1 = *(const float4*)(s + 8 * i + 4);
        P8& d = i ? ahi : alo;
        d.h[0] = (__bf16)f0.x; d.h[1] = (__bf16)f0.y;
        d.h[2] = (__bf16)f0.z; d.h[3] = (__bf16)f0.w;
        d.h[4] = (__bf16)f1.x; d.h[5] = (__bf16)f1.y;
        d.h[6] = (__bf16)f1.z; d.h[7] = (__bf16)f1.w;
      }
    } else {
      const __bf16* s = Ab + aBase + (size_t)(m0 + arow) * lda + kk + akb;
      alo.q = *(const uint4*)s;
      ahi.q = *(const uint4*)(s + 8);
    }
    const __bf16* s = b_trans
        ? (Bp + bBase + (size_t)(n0 + brow) * ldb + kk + bkb)
        : (Bp + bBase + (size_t)(kk + krow) * ldb + n0 + nb);
    blo.q = *(const uint4*)s;
    bhi.q = *(const uint4*)(s + 8);
  };

  // ---- regs -> LDS buffer ----
  auto store_lds = [&](int buf, P8& alo, P8& ahi, P8& blo, P8& bhi) {
    *(uint4*)&As[buf][arow * LDSS + akb]     = alo.q;
    *(uint4*)&As[buf][arow * LDSS + akb + 8] = ahi.q;
    if (b_trans) {
      *(uint4*)&Bt[buf][brow * LDSS + bkb]     = blo.q;
      *(uint4*)&Bt[buf][brow * LDSS + bkb + 8] = bhi.q;
    } else {
#pragma unroll
      for (int i = 0; i < 8; ++i) Bt[buf][(nb + i) * LDSS + krow]     = blo.h[i];
#pragma unroll
      for (int i = 0; i < 8; ++i) Bt[buf][(nb + 8 + i) * LDSS + krow] = bhi.h[i];
    }
  };

  // prologue: stage K-slab 0 into buffer 0
  {
    P8 alo, ahi, blo, bhi;
    load_regs(0, alo, ahi, blo, bhi);
    store_lds(0, alo, ahi, blo, bhi);
  }
  __syncthreads();

  const int aoff1 = lhalf ? 4 : 0;    // A frag uint offsets (K/2)
  const int aoff2 = lhalf ? 12 : 8;
  const int boff1 = lhalf ? 4 : 0;    // B frag uint offsets (same K order as A)
  const int boff2 = lhalf ? 12 : 8;

  int cur = 0;
  for (int k0 = 0; k0 < K; k0 += TK, cur ^= 1) {
    const int kn = k0 + TK;
    const bool has_next = kn < K;
    P8 alo, ahi, blo, bhi;
    if (has_next) {
      load_regs(kn, alo, ahi, blo, bhi);    // retires in the shadow of wmma
      if (kn + TK < K) {                    // keep L2->WGP pipeline ahead
        const char* pa = a_f32
            ? (const char*)(Af + aBase + (size_t)(m0 + arow) * lda + kn + TK + akb)
            : (const char*)(Ab + aBase + (size_t)(m0 + arow) * lda + kn + TK + akb);
        const char* pb = b_trans
            ? (const char*)(Bp + bBase + (size_t)(n0 + brow) * ldb + kn + TK + bkb)
            : (const char*)(Bp + bBase + (size_t)(kn + TK + krow) * ldb + n0 + nb);
        __builtin_prefetch(pa, 0, 3);
        __builtin_prefetch(pb, 0, 3);
      }
    }

    // ---- fragments per ISA 16-bit layouts ----
    // A 16x32: lanes 0-15 hold K{0-7,16-23}, lanes 16-31 hold K{8-15,24-31}
    F a[2], bfr[4];
    const unsigned* ar0 = (const unsigned*)&As[cur][(miw + l16) * LDSS];
    const unsigned* ar1 = (const unsigned*)&As[cur][(miw + 16 + l16) * LDSS];
#pragma unroll
    for (int j = 0; j < 4; ++j) {
      a[0].u[j]     = ar0[aoff1 + j];
      a[0].u[4 + j] = ar0[aoff2 + j];
      a[1].u[j]     = ar1[aoff1 + j];
      a[1].u[4 + j] = ar1[aoff2 + j];
    }
    // B 32x16: lane = N, lanes 0-15 hold K{0-7,16-23}, lanes 16-31 hold K{8-15,24-31}
#pragma unroll
    for (int t = 0; t < 4; ++t) {
      const unsigned* br = (const unsigned*)&Bt[cur][(niw + 16 * t + l16) * LDSS];
#pragma unroll
      for (int j = 0; j < 4; ++j) { bfr[t].u[j] = br[boff1 + j]; bfr[t].u[4 + j] = br[boff2 + j]; }
    }

#pragma unroll
    for (int in = 0; in < 4; ++in) {
      acc[0][in].v = __builtin_amdgcn_wmma_f32_16x16x32_bf16(
          false, a[0].v, false, bfr[in].v, (short)0, acc[0][in].v, false, false);
      acc[1][in].v = __builtin_amdgcn_wmma_f32_16x16x32_bf16(
          false, a[1].v, false, bfr[in].v, (short)0, acc[1][in].v, false, false);
    }

    if (has_next) store_lds(cur ^ 1, alo, ahi, blo, bhi);
    __syncthreads();
  }

  // ---- epilogue; C/D layout: VGPR r -> M = r + (lane>=16 ? 8 : 0), N = lane&15
#pragma unroll
  for (int im = 0; im < 2; ++im)
#pragma unroll
    for (int in = 0; in < 4; ++in) {
      int gm_base = m0 + miw + im * 16 + lhalf * 8;
      int gn      = n0 + niw + in * 16 + l16;
#pragma unroll
      for (int r = 0; r < 8; ++r) {
        int gm = gm_base + r;
        float val = acc[im][in].f[r] * scale;
        if (bias) val += bias[gm];
        if (res)  val += res[(size_t)b * sRb + (size_t)gm * ldr + gn];
        size_t idx = d_trans ? ((size_t)gn * ldo + gm)
                             : ((size_t)gm * ldo + gn);
        idx += (size_t)b * sDb;
        if (d_bf16) ((__bf16*)Dp)[idx] = (__bf16)val;
        else        ((float*)Dp)[idx]  = val;
      }
    }
}

// ---------------------------------------------------------------------------
extern "C" void kernel_launch(void* const* d_in, const int* in_sizes, int n_in,
                              void* d_out, int out_size, void* d_ws, size_t ws_size,
                              hipStream_t stream) {
  (void)in_sizes; (void)n_in; (void)out_size; (void)ws_size;
  const float* x     = (const float*)d_in[0];
  const float* gamma = (const float*)d_in[1];
  const float* beta  = (const float*)d_in[2];
  const float* Wq = (const float*)d_in[3]; const float* bq = (const float*)d_in[4];
  const float* Wk = (const float*)d_in[5]; const float* bk = (const float*)d_in[6];
  const float* Wv = (const float*)d_in[7]; const float* bv = (const float*)d_in[8];
  const float* Wp = (const float*)d_in[9]; const float* bp = (const float*)d_in[10];
  float* out = (float*)d_out;

  const int B = 16, C = 512, T = 1024;
  const long CT = (long)C * T;

  char* ws = (char*)d_ws;
  float*  stats = (float*)ws;               size_t off = 4096;
  __bf16* hn = (__bf16*)(ws + off);         off += (size_t)B * CT * 2;  // 16MB
  __bf16* q  = (__bf16*)(ws + off);         off += (size_t)B * CT * 2;  // [B,T,C]
  __bf16* kk = (__bf16*)(ws + off);         off += (size_t)B * CT * 2;
  __bf16* v  = (__bf16*)(ws + off);         off += (size_t)B * CT * 2;
  float*  sc = (float*)(ws + off);          // [B,T,S] f32, 64MB; probs reuse it

  gn_stats_kernel<<<B * 32, 256, 0, stream>>>(x, stats);
  gn_apply_kernel<<<(B * C * T) / 1024, 256, 0, stream>>>(x, stats, gamma, beta, hn);

  dim3 blk(256);
  dim3 gq(T / TN, C / TM, B);
  // q stored transposed [B,T,C] so scores GEMM can consume it row-major as A
  gemm_wmma_kernel<<<gq, blk, 0, stream>>>(Wq, 1, 0, C, hn, 0, CT, T,
      q, 1, 1, CT, C, bq, nullptr, 0, 0, 1.f, C);
  gemm_wmma_kernel<<<gq, blk, 0, stream>>>(Wk, 1, 0, C, hn, 0, CT, T,
      kk, 1, 0, CT, T, bk, nullptr, 0, 0, 1.f, C);
  gemm_wmma_kernel<<<gq, blk, 0, stream>>>(Wv, 1, 0, C, hn, 0, CT, T,
      v, 1, 0, CT, T, bv, nullptr, 0, 0, 1.f, C);

  // scores[b,t,s] = C^-0.5 * q^T k
  dim3 gs(T / TN, T / TM, B);
  gemm_wmma_kernel<<<gs, blk, 0, stream>>>(q, 0, (long)T * C, C, kk, 0, CT, T,
      sc, 0, 0, (long)T * T, T, nullptr, nullptr, 0, 0,
      0.044194173824159216f, C);

  softmax_kernel<<<B * T, 256, 0, stream>>>(sc);

  // PV: attn[b,c,t] = sum_s v[b,c,s] * P[b,t,s]; P = bf16 rows inside sc,
  // row stride 2*T bf16 elems. Reuse hn as attn output.
  dim3 gp(T / TN, C / TM, B);
  gemm_wmma_kernel<<<gp, blk, 0, stream>>>(v, 0, CT, T,
      (const __bf16*)sc, 1, (long)T * 2 * T, 2 * T,
      hn, 1, 0, CT, T, nullptr, nullptr, 0, 0, 1.f, T);

  // proj + bias + residual -> d_out (f32)
  gemm_wmma_kernel<<<gp, blk, 0, stream>>>(Wp, 1, 0, C, hn, 0, CT, T,
      out, 0, 0, CT, T, bp, x, CT, T, 1.f, C);
}
